// GATNet_5111011083068
// MI455X (gfx1250) — matrix operand order repaired and checked by reading, not compiled, not run on hardware
//
#include <hip/hip_runtime.h>
#include <hip/hip_bf16.h>
#include <math.h>
#include <stdint.h>

// Problem constants (from reference)
#define NN   50000
#define EE   800000
#define FIN  256
#define CCH  128
#define HC   256              // H*C
#define TOT_EDGES (EE + NN)   // edges + self loops
#define PSTR 264              // padded LDS panel row stride (bf16 units), 528 B, 16B-aligned

typedef __attribute__((ext_vector_type(16))) __bf16        bf16x16;
typedef __attribute__((ext_vector_type(8)))  float         f32x8;
typedef __attribute__((ext_vector_type(4)))  unsigned int  u32x4;

union Frag16 { bf16x16 v; u32x4 u[2]; };

// Async global -> LDS (CDNA5, ASYNCcnt-tracked). 16 bytes per enabled lane.
__device__ __forceinline__ void async_b128(unsigned lds_off, const void* gaddr) {
    asm volatile("global_load_async_to_lds_b128 %0, %1, off"
                 :: "v"(lds_off), "v"((unsigned long long)(uintptr_t)gaddr)
                 : "memory");
}
__device__ __forceinline__ void wait_async0() {
    asm volatile("s_wait_asynccnt 0" ::: "memory");
}
// Low 32 bits of a flat LDS address == LDS byte offset (aperture layout).
__device__ __forceinline__ unsigned lds_off32(const void* p) {
    return (unsigned)(uintptr_t)p;
}

// ---------------------------------------------------------------------------
// f32 -> bf16 convert (grid stride)
// ---------------------------------------------------------------------------
__global__ void f2bf_kernel(const float* __restrict__ src, unsigned short* __restrict__ dst, int n) {
    int i = blockIdx.x * blockDim.x + threadIdx.x;
    int stride = gridDim.x * blockDim.x;
    for (; i < n; i += stride)
        ((__bf16*)dst)[i] = (__bf16)src[i];
}

// ---------------------------------------------------------------------------
// eb[n,k] = bf16( elu( agg[n,k] + bias[k] ) )   (pre-pass for GEMM2)
// ---------------------------------------------------------------------------
__global__ void elu_bias_bf16(const float* __restrict__ agg, const float* __restrict__ bias,
                              unsigned short* __restrict__ eb) {
    int i = blockIdx.x * blockDim.x + threadIdx.x;
    int stride = gridDim.x * blockDim.x;
    const int n = NN * HC;
    for (; i < n; i += stride) {
        float v = agg[i] + bias[i & (HC - 1)];
        v = v > 0.f ? v : (expf(v) - 1.f);
        ((__bf16*)eb)[i] = (__bf16)v;
    }
}

// ---------------------------------------------------------------------------
// Unified WMMA GEMM:  C[M,Ntot] = A[M,256] @ B[Ntot,256]^T  (+ bvec[n] if set)
// bf16 in / f32 out.  Block = 256 thr = 8 waves; 64(M) x 64(N) tile.
// Whole 64x256 A and B panels are async-DMA'd into LDS once, then the K loop
// runs barrier-free: ds_load_b128 fragments + v_wmma_f32_16x16x32_bf16.
// ---------------------------------------------------------------------------
__global__ __launch_bounds__(256) void gemm_bf16(const unsigned short* __restrict__ A,
                                                 const unsigned short* __restrict__ B,
                                                 float* __restrict__ Cmat,
                                                 const float* __restrict__ bvec,
                                                 int M, int Ntot) {
    __shared__ __align__(16) unsigned short As[64 * PSTR];
    __shared__ __align__(16) unsigned short Bs[64 * PSTR];
    const int tid  = threadIdx.x;
    const int m0   = blockIdx.x * 64;
    const int n0   = blockIdx.y * 64;
    const int lane = tid & 31;
    const int wave = tid >> 5;
    const int wm   = wave & 3;        // M subtile (16 rows)
    const int wn   = wave >> 2;       // N subtile (32 cols)
    const int lr   = tid >> 2;        // panel-load row 0..63
    const int lc   = (tid & 3) * 8;   // panel-load col base (bf16 units)
    const int half = lane >> 4;
    const int l16  = lane & 15;

    // ---- async panel prefetch: 8 x 16B chunks per thread per panel ----
    {
        const unsigned short* agp = A + (size_t)(m0 + lr) * 256 + lc;
        const unsigned short* bgp = B + (size_t)(n0 + lr) * 256 + lc;
        unsigned la = lds_off32(&As[lr * PSTR + lc]);
        unsigned lb = lds_off32(&Bs[lr * PSTR + lc]);
        const bool inA = (m0 + lr) < M;   // OOB A rows left stale: they only
                                          // feed C rows the epilogue drops.
#pragma unroll
        for (int j = 0; j < 8; ++j) {
            if (inA) async_b128(la + j * 64, agp + j * 32);
            async_b128(lb + j * 64, bgp + j * 32);
        }
        wait_async0();
        __syncthreads();
    }

    f32x8 c0 = {}; f32x8 c1 = {};
    const unsigned short* arow  = &As[(wm * 16 + l16) * PSTR];
    const unsigned short* brow0 = &Bs[(wn * 32 + l16) * PSTR];
    const unsigned short* brow1 = &Bs[(wn * 32 + 16 + l16) * PSTR];

#pragma unroll
    for (int kk = 0; kk < 8; ++kk) {
        const int k0 = kk * 32;
        Frag16 af, bf0, bf1;
        af.u[0]  = *reinterpret_cast<const u32x4*>(arow + k0 + half * 16);
        af.u[1]  = *reinterpret_cast<const u32x4*>(arow + k0 + half * 16 + 8);
        bf0.u[0] = *reinterpret_cast<const u32x4*>(brow0 + k0 + half * 16);
        bf0.u[1] = *reinterpret_cast<const u32x4*>(brow0 + k0 + half * 16 + 8);
        bf1.u[0] = *reinterpret_cast<const u32x4*>(brow1 + k0 + half * 16);
        bf1.u[1] = *reinterpret_cast<const u32x4*>(brow1 + k0 + half * 16 + 8);
        c0 = __builtin_amdgcn_wmma_f32_16x16x32_bf16(false, af.v, false, bf0.v, (short)0, c0, false, false);
        c1 = __builtin_amdgcn_wmma_f32_16x16x32_bf16(false, af.v, false, bf1.v, (short)0, c1, false, false);
    }

    const int nA = n0 + wn * 32 + l16;
    const int nB = nA + 16;
    const float bA = bvec ? bvec[nA] : 0.f;
    const float bB = bvec ? bvec[nB] : 0.f;
#pragma unroll
    for (int r = 0; r < 8; ++r) {
        int row = m0 + wm * 16 + r + 8 * half;
        if (row < M) {
            Cmat[(size_t)row * Ntot + nA] = c0[r] + bA;
            Cmat[(size_t)row * Ntot + nB] = c1[r] + bB;
        }
    }
}

// ---------------------------------------------------------------------------
// Attention dots: a_src[n,h] = <h[n,h,:], att_src[h,:]>, same for dst.
// One wave per node; lanes 0-15 -> head 0, lanes 16-31 -> head 1.
// ---------------------------------------------------------------------------
__global__ __launch_bounds__(256) void att_dots(const float* __restrict__ h,
                                                const float* __restrict__ att_s,
                                                const float* __restrict__ att_d,
                                                float* __restrict__ asrc,
                                                float* __restrict__ adst, int M) {
    int n    = blockIdx.x * 8 + (threadIdx.x >> 5);
    int lane = threadIdx.x & 31;
    if (n >= M) return;
    const float* hp = h + (size_t)n * 256 + lane * 8;
    float ps = 0.f, pd = 0.f;
#pragma unroll
    for (int j = 0; j < 8; ++j) {
        float v = hp[j];
        ps += v * att_s[lane * 8 + j];
        pd += v * att_d[lane * 8 + j];
    }
#pragma unroll
    for (int off = 8; off >= 1; off >>= 1) {
        ps += __shfl_xor(ps, off, 16);
        pd += __shfl_xor(pd, off, 16);
    }
    if ((lane & 15) == 0) {
        asrc[n * 2 + (lane >> 4)] = ps;
        adst[n * 2 + (lane >> 4)] = pd;
    }
}

// ---------------------------------------------------------------------------
// Edge pass 1: e = leakyrelu(a_src[s]+a_dst[d]); ex = exp(e); denom[d] += ex
// (max-subtraction dropped: |e| is O(1), so plain exp is numerically safe and
// mathematically identical after normalization)
// ---------------------------------------------------------------------------
__global__ void edge_pass1(const int* __restrict__ ei,
                           const float* __restrict__ asrc,
                           const float* __restrict__ adst,
                           float* __restrict__ ex,
                           float* __restrict__ denom) {
    int i = blockIdx.x * blockDim.x + threadIdx.x;
    if (i >= TOT_EDGES) return;
    int s, d;
    if (i < EE) { s = ei[i]; d = ei[EE + i]; } else { s = d = i - EE; }
    float e0 = asrc[s * 2 + 0] + adst[d * 2 + 0];
    float e1 = asrc[s * 2 + 1] + adst[d * 2 + 1];
    e0 = e0 > 0.f ? e0 : 0.2f * e0;
    e1 = e1 > 0.f ? e1 : 0.2f * e1;
    float x0 = expf(e0), x1 = expf(e1);
    ex[(size_t)i * 2 + 0] = x0;
    ex[(size_t)i * 2 + 1] = x1;
    atomicAdd(&denom[d * 2 + 0], x0);
    atomicAdd(&denom[d * 2 + 1], x1);
}

// ---------------------------------------------------------------------------
// Edge pass 2: agg[d] += alpha * h[s].  One wave per edge; 8 f32 per lane.
// h and agg (51 MB each) stay L2-resident (192 MB L2) -> L2-bandwidth bound.
// ---------------------------------------------------------------------------
__global__ __launch_bounds__(256) void edge_pass2(const int* __restrict__ ei,
                                                  const float* __restrict__ h,
                                                  const float* __restrict__ ex,
                                                  const float* __restrict__ denom,
                                                  float* __restrict__ agg) {
    long long i = (long long)blockIdx.x * 8 + (threadIdx.x >> 5);
    if (i >= TOT_EDGES) return;
    int lane = threadIdx.x & 31;
    int s, d;
    if (i < EE) { s = ei[i]; d = ei[EE + i]; } else { s = d = (int)(i - EE); }
    int head = lane >> 4;   // elems lane*8..+8 lie entirely in head = lane>>4
    float alpha = ex[(size_t)i * 2 + head] / fmaxf(denom[d * 2 + head], 1e-16f);
    const float* hp = h + (size_t)s * 256 + lane * 8;
    float*       op = agg + (size_t)d * 256 + lane * 8;
    float4 v0 = *reinterpret_cast<const float4*>(hp);
    float4 v1 = *reinterpret_cast<const float4*>(hp + 4);
    atomicAdd(op + 0, v0.x * alpha);
    atomicAdd(op + 1, v0.y * alpha);
    atomicAdd(op + 2, v0.z * alpha);
    atomicAdd(op + 3, v0.w * alpha);
    atomicAdd(op + 4, v1.x * alpha);
    atomicAdd(op + 5, v1.y * alpha);
    atomicAdd(op + 6, v1.z * alpha);
    atomicAdd(op + 7, v1.w * alpha);
}

// ---------------------------------------------------------------------------
// Launch
// ---------------------------------------------------------------------------
extern "C" void kernel_launch(void* const* d_in, const int* in_sizes, int n_in,
                              void* d_out, int out_size, void* d_ws, size_t ws_size,
                              hipStream_t stream) {
    const float* x       = (const float*)d_in[0];
    const int*   ei      = (const int*)d_in[1];   // [2, E] (src row then dst row)
    const float* W       = (const float*)d_in[2]; // [256,256]
    const float* att_src = (const float*)d_in[3]; // [2,128]
    const float* att_dst = (const float*)d_in[4];
    const float* bias    = (const float*)d_in[5]; // [256]
    const float* lin_w   = (const float*)d_in[6]; // [128,256]
    const float* lin_b   = (const float*)d_in[7]; // [128]
    float* y = (float*)d_out;

    char*  ws  = (char*)d_ws;
    size_t off = 0;
    auto carve = [&](size_t bytes) -> char* {
        char* p = ws + off;
        off = (off + bytes + 255) & ~(size_t)255;
        return p;
    };
    unsigned short* xb  = (unsigned short*)carve((size_t)NN * FIN * 2);
    unsigned short* Wb  = (unsigned short*)carve((size_t)HC * FIN * 2);
    unsigned short* Lb  = (unsigned short*)carve((size_t)CCH * HC * 2);
    unsigned short* eb  = (unsigned short*)carve((size_t)NN * HC * 2);
    float* h     = (float*)carve((size_t)NN * HC * 4);
    float* asrc  = (float*)carve((size_t)NN * 2 * 4);
    float* adst  = (float*)carve((size_t)NN * 2 * 4);
    float* exbuf = (float*)carve((size_t)TOT_EDGES * 2 * 4);
    float* denom = (float*)carve((size_t)NN * 2 * 4);
    float* agg   = (float*)carve((size_t)NN * HC * 4);

    // 1) bf16 converts
    f2bf_kernel<<<(NN * FIN + 1023) / 1024, 1024, 0, stream>>>(x, xb, NN * FIN);
    f2bf_kernel<<<(HC * FIN + 255) / 256, 256, 0, stream>>>(W, Wb, HC * FIN);
    f2bf_kernel<<<(CCH * HC + 255) / 256, 256, 0, stream>>>(lin_w, Lb, CCH * HC);

    // 2) h = x @ W^T  (async-prefetch WMMA)
    gemm_bf16<<<dim3((NN + 63) / 64, HC / 64), 256, 0, stream>>>(xb, Wb, h, nullptr, NN, HC);

    // 3) attention dots
    att_dots<<<(NN + 7) / 8, 256, 0, stream>>>(h, att_src, att_dst, asrc, adst, NN);

    // 4) zero accumulators (must re-zero every call)
    hipMemsetAsync(denom, 0, (size_t)NN * 2 * 4, stream);
    hipMemsetAsync(agg, 0, (size_t)NN * HC * 4, stream);

    // 5) edge softmax denominators
    edge_pass1<<<(TOT_EDGES + 255) / 256, 256, 0, stream>>>(ei, asrc, adst, exbuf, denom);

    // 6) alpha-weighted scatter-add of messages
    edge_pass2<<<(TOT_EDGES + 7) / 8, 256, 0, stream>>>(ei, h, exbuf, denom, agg);

    // 7) eb = bf16(elu(agg + bias));  y = eb @ lin_w^T + lin_b
    elu_bias_bf16<<<(NN * HC + 1023) / 1024, 1024, 0, stream>>>(agg, bias, eb);
    gemm_bf16<<<dim3((NN + 63) / 64, CCH / 64), 256, 0, stream>>>(eb, Lb, y, lin_b, NN, CCH);
}
